// HybridQLSTM_65481071399972
// MI455X (gfx1250) — matrix operand order repaired and checked by reading, not compiled, not run on hardware
//
#include <hip/hip_runtime.h>
#include <cstdint>

// ---------------------------------------------------------------------------
// HybridQLSTM for MI455X (gfx1250)
//   - fused bf16 WMMA GEMMs (Wq|Wu|Wo fused N=3072; enc_f|enc_i fused N=2048)
//   - GLOBAL_LOAD_ASYNC_TO_LDS_B128 double-buffered staging (ASYNCcnt)
//   - exact low-rank correction for the sequential quantum-wire updates
// T=128, B=256, DIN=1024, H=QDIM=1024, D=2048, NQ=8
// ---------------------------------------------------------------------------

#define TT   128
#define BB   256
#define DIN_ 1024
#define HH   1024
#define DD   2048
#define NQ_  8

#if __has_builtin(__builtin_amdgcn_global_load_async_to_lds_b128) && \
    __has_builtin(__builtin_amdgcn_s_wait_asynccnt)
#define USE_ASYNC_LDS 1
#else
#define USE_ASYNC_LDS 0
#endif

typedef __attribute__((ext_vector_type(16))) __bf16 v16bf;
typedef __attribute__((ext_vector_type(8)))  float  v8f;
typedef __attribute__((ext_vector_type(4)))  int    v4i;

typedef __attribute__((address_space(1))) v4i* gptr_v4i;   // global int4*
typedef __attribute__((address_space(3))) v4i* lptr_v4i;   // LDS int4*

union FragBF { v16bf v; uint4 q[2]; };

__device__ __forceinline__ unsigned short f2bf(float f) {
    unsigned u = __float_as_uint(f);
    unsigned r = u + 0x7FFFu + ((u >> 16) & 1u);   // round-to-nearest-even
    return (unsigned short)(r >> 16);
}
__device__ __forceinline__ float sigm(float x) { return 1.0f / (1.0f + __expf(-x)); }

// --------------------------- small helper kernels ---------------------------

__global__ void k_zero_f32(float* p, int n) {
    int i = blockIdx.x * blockDim.x + threadIdx.x;
    if (i < n) p[i] = 0.0f;
}

// zero the h-half of comb_bf: comb[row*DD + DIN + col] = 0
__global__ void k_zero_h(unsigned short* comb) {
    int i = blockIdx.x * blockDim.x + threadIdx.x;   // 0 .. BB*HH-1
    int row = i >> 10, col = i & 1023;
    comb[row * DD + DIN_ + col] = 0;
}

__global__ void k_cvt(const float* __restrict__ src, unsigned short* __restrict__ dst, int n) {
    int i = blockIdx.x * blockDim.x + threadIdx.x;
    if (i < n) dst[i] = f2bf(src[i]);
}

__global__ void k_concat_bias(const float* __restrict__ bq, const float* __restrict__ bu,
                              const float* __restrict__ bo, float* __restrict__ bcat) {
    int i = blockIdx.x * blockDim.x + threadIdx.x;   // 0 .. 3071
    float v = (i < HH) ? bq[i] : (i < 2 * HH) ? bu[i - HH] : bo[i - 2 * HH];
    bcat[i] = v;
}

// x half of comb (h half is written by the wires kernel each step)
__global__ void k_pack_x(const float* __restrict__ x, unsigned short* __restrict__ comb) {
    int i = blockIdx.x * blockDim.x + threadIdx.x;   // 0 .. BB*DIN-1
    int row = i >> 10, col = i & 1023;
    comb[row * DD + col] = f2bf(x[i]);
}

// ------------------------------- WMMA GEMM ---------------------------------
// C[M,N] = A[M,K] * W[N,K]^T (+bias[N]); A,W bf16 row-major, C f32.
// Optionally side-writes bf16 of columns [0,bfN) into Cbf[M,bfN].
// 256 threads = 8 wave32s (2Mx4N); block tile 64x128; wave tile 32x32 =
// 2x2 v_wmma_f32_16x16x32_bf16 accumulators; double-buffered async LDS staging.
#define AS_STRIDE 40   // 32 bf16 + 8 pad -> 80B rows (conflict-free 16B lanes)
#define BS_STRIDE 40

__global__ void __launch_bounds__(256)
k_gemm_bf16(const unsigned short* __restrict__ A,
            const unsigned short* __restrict__ W,
            const float* __restrict__ bias,
            float* __restrict__ C,
            unsigned short* __restrict__ Cbf, int bfN,
            int M, int N, int K) {
    __shared__ unsigned short As[2][64 * AS_STRIDE];
    __shared__ unsigned short Bs[2][128 * BS_STRIDE];

    const int tid  = threadIdx.x;
    const int lane = tid & 31;
    const int wave = tid >> 5;
    const int wm   = wave >> 2;        // 0..1
    const int wn   = wave & 3;         // 0..3
    const int lr   = lane & 15;
    const int lh   = lane >> 4;        // lane half (wave32)

    const int nBase = blockIdx.x * 128;
    const int mBase = blockIdx.y * 64;

    const int ar = tid >> 2;           // 0..63
    const int ac = (tid & 3) * 8;      // 0,8,16,24
    const int br0 = tid >> 2,        bc0 = (tid & 3) * 8;
    const int br1 = (tid + 256) >> 2, bc1 = ((tid + 256) & 3) * 8;

    const unsigned short* gA  = A + (size_t)(mBase + ar) * K + ac;
    const unsigned short* gB0 = W + (size_t)(nBase + br0) * K + bc0;
    const unsigned short* gB1 = W + (size_t)(nBase + br1) * K + bc1;

    v8f acc[2][2] = {};

    const int nk = K >> 5;

    auto stage = [&](int k0, int s) {
#if USE_ASYNC_LDS
        __builtin_amdgcn_global_load_async_to_lds_b128(
            (gptr_v4i)(gA + k0),
            (lptr_v4i)&As[s][ar * AS_STRIDE + ac], 0, 0);
        __builtin_amdgcn_global_load_async_to_lds_b128(
            (gptr_v4i)(gB0 + k0),
            (lptr_v4i)&Bs[s][br0 * BS_STRIDE + bc0], 0, 0);
        __builtin_amdgcn_global_load_async_to_lds_b128(
            (gptr_v4i)(gB1 + k0),
            (lptr_v4i)&Bs[s][br1 * BS_STRIDE + bc1], 0, 0);
#else
        *reinterpret_cast<uint4*>(&As[s][ar * AS_STRIDE + ac]) =
            *reinterpret_cast<const uint4*>(gA + k0);
        *reinterpret_cast<uint4*>(&Bs[s][br0 * BS_STRIDE + bc0]) =
            *reinterpret_cast<const uint4*>(gB0 + k0);
        *reinterpret_cast<uint4*>(&Bs[s][br1 * BS_STRIDE + bc1]) =
            *reinterpret_cast<const uint4*>(gB1 + k0);
        if (k0 + 32 < K) {
            __builtin_prefetch(gA + k0 + 32, 0, 3);
            __builtin_prefetch(gB0 + k0 + 32, 0, 3);
        }
#endif
    };

    stage(0, 0);
#if USE_ASYNC_LDS
    __builtin_amdgcn_s_wait_asynccnt(0);
#endif
    __syncthreads();

    for (int k = 0; k < nk; ++k) {
        const int cur = k & 1;
        if (k + 1 < nk) stage((k + 1) << 5, cur ^ 1);

        // A fragment (16-bit A layout: lane half lh holds K blocks {lh*8, lh*8+16})
        FragBF a[2], b[2];
        #pragma unroll
        for (int mt = 0; mt < 2; ++mt) {
            const unsigned short* ap = &As[cur][(wm * 32 + mt * 16 + lr) * AS_STRIDE];
            a[mt].q[0] = *reinterpret_cast<const uint4*>(ap + lh * 8);
            a[mt].q[1] = *reinterpret_cast<const uint4*>(ap + lh * 8 + 16);
        }
        // B fragment (lane n holds B[:,n] = W[n, K blocks {lh*8, lh*8+16}])
        #pragma unroll
        for (int nt = 0; nt < 2; ++nt) {
            const unsigned short* bp = &Bs[cur][(wn * 32 + nt * 16 + lr) * BS_STRIDE + lh * 8];
            b[nt].q[0] = *reinterpret_cast<const uint4*>(bp);
            b[nt].q[1] = *reinterpret_cast<const uint4*>(bp + 16);
        }
        #pragma unroll
        for (int mt = 0; mt < 2; ++mt)
            #pragma unroll
            for (int nt = 0; nt < 2; ++nt)
                acc[mt][nt] = __builtin_amdgcn_wmma_f32_16x16x32_bf16(
                    false, a[mt].v, false, b[nt].v, (short)0, acc[mt][nt], false, false);

        if (k + 1 < nk) {
#if USE_ASYNC_LDS
            __builtin_amdgcn_s_wait_asynccnt(0);
#endif
            __syncthreads();
        }
    }

    // epilogue: C/D f32 layout (VGPR r: lanes0-15 M=r, lanes16-31 M=r+8)
    #pragma unroll
    for (int mt = 0; mt < 2; ++mt)
        #pragma unroll
        for (int nt = 0; nt < 2; ++nt) {
            int col = nBase + wn * 32 + nt * 16 + lr;
            float bv = bias ? bias[col] : 0.0f;
            #pragma unroll
            for (int r = 0; r < 8; ++r) {
                int row = mBase + wm * 32 + mt * 16 + lh * 8 + r;
                float v = acc[mt][nt][r] + bv;
                C[(size_t)row * N + col] = v;
                if (Cbf && col < bfN) Cbf[(size_t)row * bfN + col] = f2bf(v);
            }
        }
}

// ------------------- fused wires + LSTM pointwise update -------------------
// One block per batch row. Wave w reduces base dots s_w (f) and s_{w+8} (i)
// with shfl; thread 0 applies the exact rank correction for the sequential
// wires (only cols<8 change):  dot_i = s_i + sum_{j<i}(v_j - z0[j]) w_i[j].
__global__ void __launch_bounds__(256)
k_wires_pointwise(const float* __restrict__ C1,   // [B, 3072] : zq | gpre | opre
                  const float* __restrict__ C2,   // [B, 2048] : zf0 | zi0
                  const float* __restrict__ wf_w, const float* __restrict__ wf_b,
                  const float* __restrict__ wi_w, const float* __restrict__ wi_b,
                  float* __restrict__ cx,
                  unsigned short* __restrict__ comb,   // bf16 [B, D]; writes h half
                  float* __restrict__ out_t,
                  float* __restrict__ out_h, float* __restrict__ out_c) {
    const int row  = blockIdx.x;
    const int tid  = threadIdx.x;
    const int wave = tid >> 5;
    const int lane = tid & 31;

    const float* zf = C2 + (size_t)row * 2048;
    const float* zi = zf + HH;
    const float* gp = C1 + (size_t)row * 3072 + HH;
    const float* op = gp + HH;

    __shared__ float sdot[16];
    __shared__ float vf[8], vi[8];

    #pragma unroll
    for (int h = 0; h < 2; ++h) {
        const float* z = h ? zi : zf;
        const float* w = (h ? wi_w : wf_w) + wave * HH;
        float p = 0.0f;
        for (int n = lane; n < HH; n += 32) p += z[n] * w[n];
        #pragma unroll
        for (int s = 16; s > 0; s >>= 1) p += __shfl_down(p, s);
        if (lane == 0) sdot[wave + h * 8] = p;
    }
    __syncthreads();

    if (tid == 0) {
        float v[8];
        for (int i = 0; i < NQ_; ++i) {
            float d = sdot[i] + wf_b[i];
            for (int j = 0; j < i; ++j) d += (v[j] - zf[j]) * wf_w[i * HH + j];
            v[i] = tanhf(d);
            vf[i] = v[i];
        }
        for (int i = 0; i < NQ_; ++i) {
            float d = sdot[8 + i] + wi_b[i];
            for (int j = 0; j < i; ++j) d += (v[j] - zi[j]) * wi_w[i * HH + j];
            v[i] = tanhf(d);
            vi[i] = v[i];
        }
    }
    __syncthreads();

    const size_t base = (size_t)row * HH;
    for (int n = tid; n < HH; n += 256) {
        float fz = (n < NQ_) ? vf[n] : zf[n];
        float iz = (n < NQ_) ? vi[n] : zi[n];
        float f  = sigm(fz);
        float ig = sigm(iz);
        float g  = tanhf(gp[n]);
        float o  = sigm(op[n]);
        float c  = f * cx[base + n] + ig * g;
        float hv = o * tanhf(c);
        cx[base + n] = c;
        comb[(size_t)row * DD + DIN_ + n] = f2bf(hv);   // next step's h input
        out_t[base + n] = hv;
        out_h[base + n] = hv;
        out_c[base + n] = c;
    }
}

// --------------------------------- driver ----------------------------------

extern "C" void kernel_launch(void* const* d_in, const int* in_sizes, int n_in,
                              void* d_out, int out_size, void* d_ws, size_t ws_size,
                              hipStream_t stream) {
    (void)in_sizes; (void)n_in; (void)out_size; (void)ws_size;

    const float* x_all = (const float*)d_in[0];   // [T,B,DIN]
    const float* Wq    = (const float*)d_in[1];   // [H, D]
    const float* bq    = (const float*)d_in[2];
    const float* enc_f = (const float*)d_in[3];   // [H, H]
    const float* wf_w  = (const float*)d_in[4];   // [NQ, H]
    const float* wf_b  = (const float*)d_in[5];
    const float* enc_i = (const float*)d_in[6];
    const float* wi_w  = (const float*)d_in[7];
    const float* wi_b  = (const float*)d_in[8];
    const float* Wu    = (const float*)d_in[9];
    const float* bu    = (const float*)d_in[10];
    const float* Wo    = (const float*)d_in[11];
    const float* bo    = (const float*)d_in[12];

    float* out   = (float*)d_out;                 // outs | hx | cx
    float* out_h = out + (size_t)TT * BB * HH;
    float* out_c = out_h + (size_t)BB * HH;

    // ---- workspace carve-out ----
    char* ws = (char*)d_ws;
    auto takeU = [&](size_t elems) { unsigned short* p = (unsigned short*)ws; ws += elems * 2; return p; };
    auto takeF = [&](size_t elems) { float* p = (float*)ws; ws += elems * 4; return p; };

    unsigned short* Wcat_bf  = takeU((size_t)3 * HH * DD);   // [3072, 2048] Wq|Wu|Wo
    unsigned short* Ecat_bf  = takeU((size_t)2 * HH * HH);   // [2048, 1024] enc_f|enc_i
    unsigned short* comb_bf  = takeU((size_t)BB * DD);       // [B, D]
    unsigned short* zq_bf    = takeU((size_t)BB * HH);       // [B, H]
    float* bcat = takeF(3 * HH);                             // bq|bu|bo
    float* cx   = takeF((size_t)BB * HH);
    float* C1   = takeF((size_t)BB * 3 * HH);                // zq | gpre | opre
    float* C2   = takeF((size_t)BB * 2 * HH);                // zf0 | zi0

    // ---- one-time (per call) weight prep ----
    k_cvt<<<(HH * DD) / 256, 256, 0, stream>>>(Wq, Wcat_bf,                       HH * DD);
    k_cvt<<<(HH * DD) / 256, 256, 0, stream>>>(Wu, Wcat_bf + (size_t)HH * DD,     HH * DD);
    k_cvt<<<(HH * DD) / 256, 256, 0, stream>>>(Wo, Wcat_bf + (size_t)2 * HH * DD, HH * DD);
    k_cvt<<<(HH * HH) / 256, 256, 0, stream>>>(enc_f, Ecat_bf,                    HH * HH);
    k_cvt<<<(HH * HH) / 256, 256, 0, stream>>>(enc_i, Ecat_bf + (size_t)HH * HH,  HH * HH);
    k_concat_bias<<<(3 * HH) / 256, 256, 0, stream>>>(bq, bu, bo, bcat);
    k_zero_f32<<<(BB * HH) / 256, 256, 0, stream>>>(cx, BB * HH);
    k_zero_h<<<(BB * HH) / 256, 256, 0, stream>>>(comb_bf);

    const dim3 g1(3 * HH / 128, BB / 64);   // (24, 4) = 96 blocks
    const dim3 g2(2 * HH / 128, BB / 64);   // (16, 4) = 64 blocks

    for (int t = 0; t < TT; ++t) {
        const float* xt = x_all + (size_t)t * BB * DIN_;
        k_pack_x<<<(BB * DIN_) / 256, 256, 0, stream>>>(xt, comb_bf);

        // C1 = comb @ [Wq|Wu|Wo]^T + [bq|bu|bo]; side-writes zq (cols<1024) as bf16
        k_gemm_bf16<<<g1, 256, 0, stream>>>(comb_bf, Wcat_bf, bcat, C1,
                                            zq_bf, HH, BB, 3 * HH, DD);
        // C2 = zq @ [enc_f|enc_i]^T
        k_gemm_bf16<<<g2, 256, 0, stream>>>(zq_bf, Ecat_bf, nullptr, C2,
                                            nullptr, 0, BB, 2 * HH, HH);

        k_wires_pointwise<<<BB, 256, 0, stream>>>(
            C1, C2, wf_w, wf_b, wi_w, wi_b,
            cx, comb_bf, out + (size_t)t * BB * HH, out_h, out_c);
    }
}
